// SMoE_MHA_enc_version_1_36661840839471
// MI455X (gfx1250) — matrix operand order repaired and checked by reading, not compiled, not run on hardware
//
#include <hip/hip_runtime.h>

// Problem constants (from reference)
#define B_    8
#define OUT_  512
#define S_    400
#define C_    19
#define NR_   4
#define D_    (OUT_ * C_)        // 9728  expert GEMM K
#define G_    (OUT_ * NR_ * C_)  // 38912 gating K
#define M_    (B_ * S_)          // 3200  rows

// GEMM tiling
#define TM   128
#define TN   128
#define TK   64   // LDS stage depth (two K=32 WMMA steps per stage)
#define AST  72   // padded LDS row stride (bf16 elems): 144B rows -> conflict-free, 16B aligned

typedef __attribute__((ext_vector_type(16))) __bf16       v16bf;
typedef __attribute__((ext_vector_type(2)))  __bf16       v2bf;
typedef __attribute__((ext_vector_type(8)))  float        v8f;
typedef __attribute__((ext_vector_type(4)))  unsigned int v4u;

union Frag { v16bf v; v4u q[2]; };

// pack two f32 -> packed bf16 (RNE).
// Preference: hw builtin (v_cvt_pk_bf16_f32) > native fptrunc pair > integer RNE pack.
#if __has_builtin(__builtin_amdgcn_cvt_pk_bf16_f32)
__device__ __forceinline__ unsigned int pack_bf16(float x, float y) {
  union { v2bf v; unsigned int u; } r;
  r.v = __builtin_amdgcn_cvt_pk_bf16_f32(x, y);
  return r.u;
}
#elif defined(__AMDGCN__)
__device__ __forceinline__ unsigned int pack_bf16(float x, float y) {
  union { v2bf v; unsigned int u; } r;
  r.v[0] = (__bf16)x;   // fptrunc f32->bf16 (RNE); backend packs pairs
  r.v[1] = (__bf16)y;
  return r.u;
}
#else
__device__ __forceinline__ unsigned int pack_bf16(float x, float y) {
  unsigned int ux = __float_as_uint(x);
  unsigned int uy = __float_as_uint(y);
  ux += 0x7FFFu + ((ux >> 16) & 1u);   // RNE
  uy += 0x7FFFu + ((uy >> 16) & 1u);
  return (ux >> 16) | (uy & 0xFFFF0000u);
}
#endif

// ---------------------------------------------------------------------------
// Kernel 1: gating scores  score[b,s,e] = sum_{o,r,c} x_r[b,o,s,c]*Wg[e,o*76+r*19+c] + bg[e]
// ---------------------------------------------------------------------------
__global__ __launch_bounds__(256) void gating_kernel(
    const float* __restrict__ x0, const float* __restrict__ x1,
    const float* __restrict__ x2, const float* __restrict__ x3,
    const float* __restrict__ Wg, const float* __restrict__ bg,
    float* __restrict__ scores)
{
  const int i = blockIdx.x;            // 0..3199
  const int b = i / S_;
  const int s = i - b * S_;
  const float* xs[4] = {x0, x1, x2, x3};

  float acc0 = 0.f, acc1 = 0.f, acc2 = 0.f, acc3 = 0.f;

  for (int j = threadIdx.x; j < OUT_ * C_; j += 256) {
    const int o = j / C_;
    const int c = j - o * C_;
    const size_t xoff  = ((size_t)b * OUT_ + o) * (size_t)(S_ * C_) + (size_t)s * C_ + c;
    const int    wbase = o * (NR_ * C_) + c;
#pragma unroll
    for (int r = 0; r < NR_; ++r) {
      const float xv = xs[r][xoff];
      const int wc = wbase + r * C_;
      acc0 += xv * Wg[0 * G_ + wc];
      acc1 += xv * Wg[1 * G_ + wc];
      acc2 += xv * Wg[2 * G_ + wc];
      acc3 += xv * Wg[3 * G_ + wc];
    }
  }

  __shared__ float red[4][256];
  red[0][threadIdx.x] = acc0;
  red[1][threadIdx.x] = acc1;
  red[2][threadIdx.x] = acc2;
  red[3][threadIdx.x] = acc3;
  __syncthreads();
  for (int off = 128; off > 0; off >>= 1) {
    if ((int)threadIdx.x < off) {
#pragma unroll
      for (int e = 0; e < 4; ++e)
        red[e][threadIdx.x] += red[e][threadIdx.x + off];
    }
    __syncthreads();
  }
  if (threadIdx.x < 4)
    scores[(size_t)i * NR_ + threadIdx.x] = red[threadIdx.x][0] + bg[threadIdx.x];
}

// ---------------------------------------------------------------------------
// Kernel 2: expert GEMMs (bf16 WMMA, f32 accumulate)
//   eout[ex][i][o] = X_ex[i,:] . We[ex][o,:] + be[ex][o]
// grid = (4, 25, 4), block = 256 (8 waves); wave -> 32x64 subtile (2x4 WMMA accs)
// LDS stage: 128x64 A-tile + 128x64 B-tile in bf16, 16 WMMAs per stage per wave.
// ---------------------------------------------------------------------------
__global__ __launch_bounds__(256) void expert_gemm_kernel(
    const float* __restrict__ x0, const float* __restrict__ x1,
    const float* __restrict__ x2, const float* __restrict__ x3,
    const float* __restrict__ We, const float* __restrict__ be,
    float* __restrict__ eout)
{
  const int nt = blockIdx.x;   // 0..3
  const int mt = blockIdx.y;   // 0..24
  const int ex = blockIdx.z;   // 0..3
  const float* __restrict__ X = (ex == 0) ? x0 : (ex == 1) ? x1 : (ex == 2) ? x2 : x3;
  const float* __restrict__ W = We + (size_t)ex * OUT_ * D_;
  const int i0 = mt * TM;
  const int n0 = nt * TN;

  __shared__ unsigned short As[TM * AST];
  __shared__ unsigned short Bs[TN * AST];

  const int tid   = threadIdx.x;
  const int lane  = tid & 31;
  const int w     = tid >> 5;        // wave 0..7
  const int wm    = w & 3;           // M offset /32
  const int wn    = w >> 2;          // N offset /64
  const int lrow  = lane & 15;
  const int lhalf = lane >> 4;

  v8f acc[2][4];
#pragma unroll
  for (int mi = 0; mi < 2; ++mi)
#pragma unroll
    for (int ni = 0; ni < 4; ++ni)
#pragma unroll
      for (int r = 0; r < 8; ++r) acc[mi][ni][r] = 0.f;

  for (int kk = 0; kk < D_; kk += TK) {
    // cooperative load + f32->bf16 convert into LDS (128x64 per matrix)
#pragma unroll
    for (int it = 0; it < 8; ++it) {
      const int q   = tid + it * 256;     // float4 index within 128x64 tile
      const int row = q >> 4;             // 0..127 (16 float4 per row)
      const int col = (q & 15) << 2;      // 0,4,...,60
      const float4 a   = *(const float4*)(X + (size_t)(i0 + row) * D_ + kk + col);
      const float4 bvv = *(const float4*)(W + (size_t)(n0 + row) * D_ + kk + col);
      uint2 pa, pb;
      pa.x = pack_bf16(a.x, a.y);
      pa.y = pack_bf16(a.z, a.w);
      pb.x = pack_bf16(bvv.x, bvv.y);
      pb.y = pack_bf16(bvv.z, bvv.w);
      *(uint2*)(&As[row * AST + col]) = pa;
      *(uint2*)(&Bs[row * AST + col]) = pb;
      if (it == 0) {
        // Unconditional speculative prefetch (locality 0 -> speculative TH;
        // OOB translations are silently dropped per ISA 10.5). Avoids the
        // loop-versioning a trip-count guard would cause.
        __builtin_prefetch(X + (size_t)(i0 + row) * D_ + kk + TK + col, 0, 0);
        __builtin_prefetch(W + (size_t)(n0 + row) * D_ + kk + TK + col, 0, 0);
      }
    }
    __syncthreads();

#pragma unroll
    for (int ks = 0; ks < TK; ks += 32) {
      // A fragments: lanes 0-15 (M=lane): K={0..7,16..23}; lanes 16-31: K={8..15,24..31}
      Frag fa[2];
#pragma unroll
      for (int mi = 0; mi < 2; ++mi) {
        const int m = wm * 32 + mi * 16 + lrow;
        fa[mi].q[0] = *(const v4u*)(&As[m * AST + ks + 8 * lhalf]);
        fa[mi].q[1] = *(const v4u*)(&As[m * AST + ks + 16 + 8 * lhalf]);
      }
      // B fragments: lanes 0-15 (N=lane): K={0..7,16..23}; lanes 16-31: K={8..15,24..31}
      Frag fb[4];
#pragma unroll
      for (int ni = 0; ni < 4; ++ni) {
        const int n = wn * 64 + ni * 16 + lrow;
        fb[ni].q[0] = *(const v4u*)(&Bs[n * AST + ks + 8 * lhalf]);
        fb[ni].q[1] = *(const v4u*)(&Bs[n * AST + ks + 16 + 8 * lhalf]);
      }

#pragma unroll
      for (int mi = 0; mi < 2; ++mi)
#pragma unroll
        for (int ni = 0; ni < 4; ++ni)
          acc[mi][ni] = __builtin_amdgcn_wmma_f32_16x16x32_bf16(
              false, fa[mi].v, false, fb[ni].v,
              (short)0, acc[mi][ni], false, false);
    }
    __syncthreads();
  }

  // epilogue: bias + store to workspace
  const float* __restrict__ bias = be + ex * OUT_;
  float* __restrict__ Eo = eout + (size_t)ex * M_ * OUT_;
#pragma unroll
  for (int mi = 0; mi < 2; ++mi) {
#pragma unroll
    for (int ni = 0; ni < 4; ++ni) {
      const int col = n0 + wn * 64 + ni * 16 + lrow;
      const float bv = bias[col];
#pragma unroll
      for (int r = 0; r < 8; ++r) {
        const int row = i0 + wm * 32 + mi * 16 + lhalf * 8 + r;
        Eo[(size_t)row * OUT_ + col] = acc[mi][ni][r] + bv;
      }
    }
  }
}

// ---------------------------------------------------------------------------
// Kernel 3: top-2 select + softmax + weighted combine
// ---------------------------------------------------------------------------
__global__ __launch_bounds__(256) void combine_kernel(
    const float* __restrict__ scores, const float* __restrict__ eout,
    float* __restrict__ out)
{
  const int i = blockIdx.x;            // 0..3199
  float sc[4];
#pragma unroll
  for (int e = 0; e < 4; ++e) sc[e] = scores[(size_t)i * NR_ + e];

  int e0 = 0;
#pragma unroll
  for (int e = 1; e < 4; ++e) if (sc[e] > sc[e0]) e0 = e;   // strict > : earliest on tie
  int e1 = -1;
#pragma unroll
  for (int e = 0; e < 4; ++e)
    if (e != e0 && (e1 < 0 || sc[e] > sc[e1])) e1 = e;

  const float t   = expf(sc[e1] - sc[e0]);   // sc[e0] is the max
  const float inv = 1.0f / (1.0f + t);
  const float w0  = inv;
  const float w1  = t * inv;

  const float* __restrict__ E0 = eout + ((size_t)e0 * M_ + i) * OUT_;
  const float* __restrict__ E1 = eout + ((size_t)e1 * M_ + i) * OUT_;
  for (int o = threadIdx.x; o < OUT_; o += 256)
    out[(size_t)i * OUT_ + o] = w0 * E0[o] + w1 * E1[o];
}

// ---------------------------------------------------------------------------
extern "C" void kernel_launch(void* const* d_in, const int* in_sizes, int n_in,
                              void* d_out, int out_size, void* d_ws, size_t ws_size,
                              hipStream_t stream)
{
  (void)in_sizes; (void)n_in; (void)out_size;
  const float* x0 = (const float*)d_in[0];
  const float* x1 = (const float*)d_in[1];
  const float* x2 = (const float*)d_in[2];
  const float* x3 = (const float*)d_in[3];
  const float* We = (const float*)d_in[4];
  const float* be = (const float*)d_in[5];
  const float* Wg = (const float*)d_in[6];
  const float* bg = (const float*)d_in[7];
  float* out = (float*)d_out;

  // workspace: [scores: 3200*4 floats][pad to 16384 floats][eout: 4*3200*512 floats]
  const size_t need = (16384 + (size_t)NR_ * M_ * OUT_) * sizeof(float);
  if (ws_size < need) return;
  float* scores = (float*)d_ws;
  float* eout   = (float*)d_ws + 16384;

  gating_kernel<<<M_, 256, 0, stream>>>(x0, x1, x2, x3, Wg, bg, scores);
  expert_gemm_kernel<<<dim3(OUT_ / TN, M_ / TM, NR_), 256, 0, stream>>>(
      x0, x1, x2, x3, We, be, eout);
  combine_kernel<<<M_, 256, 0, stream>>>(scores, eout, out);
}
